// GroupAttention_63041529970990
// MI455X (gfx1250) — matrix operand order repaired and checked by reading, not compiled, not run on hardware
//
#include <hip/hip_runtime.h>
#include <hip/hip_bf16.h>
#include <math.h>

// ---------------------------------------------------------------------------
// Fused Swin window attention for MI455X (gfx1250, wave32, WMMA bf16).
// One workgroup (256 thr = 8 waves) handles one (batch, window) pair:
//   LDS x-tile -> QKV GEMM -> per-head QK^T -> column softmax (query axis,
//   as in the reference) -> P@V -> output projection + bias -> global.
// Weights are read directly from global (L2-resident across 2048 blocks).
// ---------------------------------------------------------------------------

typedef __attribute__((ext_vector_type(16))) __bf16 v16bf;
typedef __attribute__((ext_vector_type(8)))  float  v8f;

union ABu { v16bf v; __bf16 e[16]; };
union CDu { v8f   v; float  e[8];  };

#define WSZ   7
#define CH    128
#define NTOK  49
#define NPAD  64
#define HEADS 4
#define HD    32
#define GRID_W 56
#define NTOT  (GRID_W*GRID_W)

__device__ __forceinline__ v8f wmma_bf16(v16bf a, v16bf b, v8f c) {
  // D = A(16x32 bf16) x B(32x16 bf16) + C(16x16 f32)
  return __builtin_amdgcn_wmma_f32_16x16x32_bf16(
      /*neg_a=*/false, a, /*neg_b=*/false, b,
      /*c_mod=*/(short)0, c, /*reuse_a=*/false, /*reuse_b=*/false);
}

// A fragment (16x32, bf16) from row-major LDS tile; ld = row stride in elems.
// Lane layout (matches B frags): lanes0-15 K=0..15; lanes16-31 K=16..31.
__device__ __forceinline__ v16bf ld_afrag_lds(const __bf16* base, int ld, int lane) {
  const int m = lane & 15, hi = lane >> 4;
  const __bf16* p0 = (const __bf16*)__builtin_assume_aligned(base + m * ld + 16 * hi, 16);
  const __bf16* p1 = (const __bf16*)__builtin_assume_aligned(base + m * ld + 8 + 16 * hi, 16);
  ABu a;
#pragma unroll
  for (int i = 0; i < 8; ++i) { a.e[i] = p0[i]; a.e[8 + i] = p1[i]; }
  return a.v;
}

// B fragment (32x16, bf16). Storage: B[k][n] laid out with contiguous k per n,
// i.e. base[n*ld + k]. Lanes0-15: K=0..15, lanes16-31: K=16..31, N = lane&15.
__device__ __forceinline__ v16bf ld_bfrag_lds(const __bf16* base, int ld, int lane) {
  const int n = lane & 15, hi = lane >> 4;
  const __bf16* p = (const __bf16*)__builtin_assume_aligned(base + n * ld + 16 * hi, 16);
  ABu b;
#pragma unroll
  for (int i = 0; i < 16; ++i) b.e[i] = p[i];
  return b.v;
}

// Same as above but sourced from a global f32 weight matrix (row-major [n][k]),
// converted to bf16 in registers. 64B contiguous per lane -> L2-friendly.
__device__ __forceinline__ v16bf ld_bfrag_gf32(const float* base, int ld, int lane) {
  const int n = lane & 15, hi = lane >> 4;
  const float* p = (const float*)__builtin_assume_aligned(base + n * ld + 16 * hi, 64);
  ABu b;
#pragma unroll
  for (int i = 0; i < 16; ++i) b.e[i] = (__bf16)p[i];
  return b.v;
}

__global__ __launch_bounds__(256) void win_attn_fused(
    const float* __restrict__ x, const float* __restrict__ qkv_w,
    const float* __restrict__ proj_w, const float* __restrict__ proj_b,
    float* __restrict__ out) {
  extern __shared__ char smem[];
  __bf16* sx  = (__bf16*)smem;            // [64][128]   x window tile
  __bf16* sq  = sx  + NPAD * CH;          // [4][64][32] Q
  __bf16* sk  = sq  + HEADS * NPAD * HD;  // [4][64][32] K
  __bf16* svt = sk  + HEADS * NPAD * HD;  // [4][32][64] V transposed
  float*  ss  = (float*)(svt + HEADS * HD * NPAD); // [2][64][64] logits (f32)
  __bf16* sp  = (__bf16*)(ss + 2 * NPAD * NPAD);   // [2][64][64] probs (bf16)
  __bf16* so  = sp + 2 * NPAD * NPAD;              // [64][128]   attn out

  const int b  = blockIdx.y;
  const int g  = blockIdx.x;            // window id 0..63
  const int gy = g >> 3, gx = g & 7;
  const int tid  = threadIdx.x;
  const int wave = tid >> 5, lane = tid & 31;
  const int hi = lane >> 4, ln = lane & 15;
  const float scale = 0.17677669529663687f; // hd^-0.5

  // ---- load x window into LDS as bf16; zero-pad tokens 49..63 -------------
  for (int idx = tid; idx < NPAD * CH; idx += 256) {
    const int t = idx >> 7, c = idx & (CH - 1);
    float v = 0.f;
    if (t < NTOK) {
      const int iy = t / WSZ, ix = t - iy * WSZ;
      const int row = (gy * WSZ + iy) * GRID_W + gx * WSZ + ix;
      v = x[((size_t)b * NTOT + row) * CH + c];
    }
    sx[idx] = (__bf16)v;
  }
  __syncthreads();

  // ---- stage 1: QKV = Xw @ qkv_w^T  (M=64, N=384, K=128) ------------------
  // 96 output tiles of 16x16, 12 per wave, 4 K-chunks each.
  for (int t = 0; t < 12; ++t) {
    const int T  = wave + 8 * t;        // 0..95
    const int mt = T & 3, nt = T >> 2;  // nt 0..23
    const int o0 = nt * 16;
    v8f acc = {};
#pragma unroll
    for (int kc = 0; kc < 4; ++kc) {
      v16bf a  = ld_afrag_lds(sx + mt * 16 * CH + kc * 32, CH, lane);
      v16bf bf = ld_bfrag_gf32(qkv_w + (size_t)o0 * CH + kc * 32, CH, lane);
      acc = wmma_bf16(a, bf, acc);
    }
    CDu cd; cd.v = acc;
    if (nt < 8) {                                     // Q
      const int head = o0 >> 5, d0 = o0 & 31;
      __bf16* dst = sq + head * NPAD * HD;
#pragma unroll
      for (int r = 0; r < 8; ++r)
        dst[(mt * 16 + r + 8 * hi) * HD + d0 + ln] = (__bf16)cd.e[r];
    } else if (nt < 16) {                             // K
      const int oo = o0 - CH, head = oo >> 5, d0 = oo & 31;
      __bf16* dst = sk + head * NPAD * HD;
#pragma unroll
      for (int r = 0; r < 8; ++r)
        dst[(mt * 16 + r + 8 * hi) * HD + d0 + ln] = (__bf16)cd.e[r];
    } else {                                          // V, stored transposed
      const int oo = o0 - 2 * CH, head = oo >> 5, d0 = oo & 31;
      __bf16* dst = svt + head * HD * NPAD;
#pragma unroll
      for (int r = 0; r < 8; ++r)
        dst[(d0 + ln) * NPAD + mt * 16 + r + 8 * hi] = (__bf16)cd.e[r];
    }
  }
  __syncthreads();

  // ---- stage 2+3: attention, two heads per pass ---------------------------
  for (int pass = 0; pass < 2; ++pass) {
    // S = Q K^T  (per head: 4x4 tiles, K=32 -> single WMMA)
#pragma unroll
    for (int i = 0; i < 4; ++i) {
      const int T  = wave + 8 * i;            // 0..31
      const int hp = T >> 4, rem = T & 15;
      const int mt = rem & 3, nt = rem >> 2;
      const int head = pass * 2 + hp;
      v16bf a  = ld_afrag_lds(sq + head * NPAD * HD + mt * 16 * HD, HD, lane);
      v16bf bf = ld_bfrag_lds(sk + head * NPAD * HD + nt * 16 * HD, HD, lane);
      v8f acc = {};
      acc = wmma_bf16(a, bf, acc);
      CDu cd; cd.v = acc;
      float* dst = ss + hp * NPAD * NPAD;
#pragma unroll
      for (int r = 0; r < 8; ++r)
        dst[(mt * 16 + r + 8 * hi) * NPAD + nt * 16 + ln] = cd.e[r];
    }
    __syncthreads();

    // softmax over the QUERY axis (axis=-2, as in the reference): one thread
    // per key-column, reduce over the 49 real query rows.
    if (tid < 2 * NPAD) {
      const int hp = tid >> 6, j = tid & (NPAD - 1);
      const float* col = ss + hp * NPAD * NPAD + j;
      float mx = -1e30f;
      for (int iq = 0; iq < NTOK; ++iq) mx = fmaxf(mx, col[iq * NPAD] * scale);
      float sum = 0.f;
      for (int iq = 0; iq < NTOK; ++iq) sum += __expf(col[iq * NPAD] * scale - mx);
      const float inv = 1.f / sum;
      __bf16* pc = sp + hp * NPAD * NPAD + j;
      for (int iq = 0; iq < NTOK; ++iq)
        pc[iq * NPAD] = (__bf16)(__expf(col[iq * NPAD] * scale - mx) * inv);
      for (int iq = NTOK; iq < NPAD; ++iq) pc[iq * NPAD] = (__bf16)0.f;
    }
    __syncthreads();

    // OUT_head = P @ V  (4x2 tiles per head, K=64 -> 2 WMMA)
#pragma unroll
    for (int i = 0; i < 2; ++i) {
      const int T  = wave + 8 * i;            // 0..15
      const int hp = T >> 3, rem = T & 7;
      const int mt = rem & 3, ntd = rem >> 2;
      const int head = pass * 2 + hp;
      v8f acc = {};
#pragma unroll
      for (int kc = 0; kc < 2; ++kc) {
        v16bf a  = ld_afrag_lds(sp + hp * NPAD * NPAD + mt * 16 * NPAD + kc * 32, NPAD, lane);
        v16bf bf = ld_bfrag_lds(svt + head * HD * NPAD + ntd * 16 * NPAD + kc * 32, NPAD, lane);
        acc = wmma_bf16(a, bf, acc);
      }
      CDu cd; cd.v = acc;
#pragma unroll
      for (int r = 0; r < 8; ++r)
        so[(mt * 16 + r + 8 * hi) * CH + head * HD + ntd * 16 + ln] = (__bf16)cd.e[r];
    }
    __syncthreads();
  }

  // ---- stage 4: projection  out = SO @ proj_w^T + proj_b ------------------
#pragma unroll
  for (int i = 0; i < 4; ++i) {
    const int T  = wave + 8 * i;        // 0..31
    const int mt = T & 3, nt = T >> 2;  // nt 0..7
    v8f acc = {};
#pragma unroll
    for (int kc = 0; kc < 4; ++kc) {
      v16bf a  = ld_afrag_lds(so + mt * 16 * CH + kc * 32, CH, lane);
      v16bf bf = ld_bfrag_gf32(proj_w + (size_t)(nt * 16) * CH + kc * 32, CH, lane);
      acc = wmma_bf16(a, bf, acc);
    }
    const float bias = proj_b[nt * 16 + ln];
    CDu cd; cd.v = acc;
#pragma unroll
    for (int r = 0; r < 8; ++r) {
      const int t2 = mt * 16 + r + 8 * hi;
      if (t2 < NTOK) {
        const int iy = t2 / WSZ, ix = t2 - iy * WSZ;
        const int row = (gy * WSZ + iy) * GRID_W + gx * WSZ + ix;
        out[((size_t)b * NTOT + row) * CH + nt * 16 + ln] = cd.e[r] + bias;
      }
    }
  }
}

extern "C" void kernel_launch(void* const* d_in, const int* in_sizes, int n_in,
                              void* d_out, int out_size, void* d_ws, size_t ws_size,
                              hipStream_t stream) {
  (void)in_sizes; (void)n_in; (void)d_ws; (void)ws_size; (void)out_size;
  const float* x      = (const float*)d_in[0];
  const float* qkv_w  = (const float*)d_in[1];
  const float* proj_w = (const float*)d_in[2];
  const float* proj_b = (const float*)d_in[3];
  // d_in[4..6] = H, W, ws scalars (56, 56, 7) — compile-time constants here.
  float* out = (float*)d_out;

  dim3 grid(64, 32);   // (windows per image, batch)
  dim3 block(256);     // 8 waves (wave32)
  const size_t lds_bytes = 131072; // 128 KB of the 320 KB/WGP
  win_attn_fused<<<grid, block, lds_bytes, stream>>>(x, qkv_w, proj_w, proj_b, out);
}
